// PairwiseTP_5377299055223
// MI455X (gfx1250) — compile-verified, matrix operand order checked by reading, not run on hardware
//
#include <hip/hip_runtime.h>

// ---------------------------------------------------------------------------
// PairwiseTP fused kernel for MI455X (gfx1250, wave32, WMMA).
//
// Shapes: E=32768 edges; radial MLP 17->64->64->768 (SiLU+LN between),
// then out[o*3+d, i*3+m] = sum_f R[o,i,f]*basis[d,m,f], out = (E,48,48) fp32.
//
// Roofline: 302MB output write => ~13us floor @ 23.3TB/s. MLP 3.5 GFLOP on
// bf16 WMMA hides under that. Everything fused; r (E x 768) never hits HBM.
// ---------------------------------------------------------------------------

typedef __attribute__((ext_vector_type(16))) __bf16 v16bf;
typedef __attribute__((ext_vector_type(8)))  float  v8f;
typedef __attribute__((ext_vector_type(4)))  unsigned int u32x4;
typedef __attribute__((ext_vector_type(4)))  float  f32x4;

union Frag { v16bf v; u32x4 q[2]; };

__device__ __forceinline__ unsigned short bf16bits(float f) {
  union { float f; unsigned u; } c; c.f = f;
  unsigned r = c.u + 0x7FFFu + ((c.u >> 16) & 1u);   // round-to-nearest-even
  return (unsigned short)(r >> 16);
}

// SiLU with hardware v_rcp_f32 (avoids IEEE div-scale/fixup sequence;
// operands already went through bf16, 1-ulp rcp is more than enough).
__device__ __forceinline__ float silu(float x) {
  return x * __builtin_amdgcn_rcpf(1.0f + __expf(-x));
}

// ---------------------------------------------------------------------------
// Prep: transpose weights to K-major bf16 in workspace so every WMMA
// B-fragment is two contiguous 16B loads per lane.
//   w1t: [64 n][32 k] (k 17..31 zero)   @ ws + 0      (2048 elems)
//   w2t: [64 n][64 k]                   @ ws + 2048   (4096 elems)
//   w3t: [768 n][64 k]                  @ ws + 6144   (49152 elems)
// ---------------------------------------------------------------------------
__global__ void prep_weights(const float* __restrict__ W1,
                             const float* __restrict__ W2,
                             const float* __restrict__ W3,
                             unsigned short* __restrict__ ws) {
  int t = blockIdx.x * 256 + threadIdx.x;
  if (t < 2048) {                         // w1t[n][k]
    int n = t >> 5, k = t & 31;
    float v = (k < 17) ? W1[k * 64 + n] : 0.0f;
    ws[t] = bf16bits(v);
  } else if (t < 2048 + 4096) {           // w2t[n][k]
    int u = t - 2048;
    int n = u >> 6, k = u & 63;
    ws[t] = bf16bits(W2[k * 64 + n]);
  } else if (t < 2048 + 4096 + 49152) {   // w3t[n][k]
    int u = t - 6144;
    int n = u >> 6, k = u & 63;
    ws[t] = bf16bits(W3[k * 768 + n]);
  }
}

// ---------------------------------------------------------------------------
// Fragment loaders. 16-bit operand layout (16x16x32):
//   lanes 0-15 : row/col = lane,    elems 0-7 -> K 0-7,  elems 8-15 -> K 16-23
//   lanes 16-31: row/col = lane-16, elems 0-7 -> K 8-15, elems 8-15 -> K 24-31
// => per lane: two contiguous 8-element (16B) chunks at K = 8*hh and 16+8*hh.
// ---------------------------------------------------------------------------
__device__ __forceinline__ Frag load_frag(const unsigned short* base,
                                          int rowcol, int stride,
                                          int koff, int hh) {
  Frag f;
  const unsigned short* p = base + rowcol * stride + koff + 8 * hh;
  f.q[0] = *(const u32x4*)(p);
  f.q[1] = *(const u32x4*)(p + 16);
  return f;
}

// LayerNorm over 64 features for 32 rows; 8 threads per row, LDS reduction.
__device__ __forceinline__ void ln_block(const float* __restrict__ hf,
                                         unsigned short* __restrict__ dst,
                                         const float* __restrict__ gamma,
                                         float* red, float* stats, int t) {
  int row = t >> 3, s = t & 7;
  float sum = 0.f, sq = 0.f;
#pragma unroll
  for (int k = 0; k < 8; ++k) {
    float x = hf[row * 64 + s * 8 + k];
    sum += x; sq += x * x;
  }
  red[t] = sum; red[256 + t] = sq;
  __syncthreads();
  if (s == 0) {
    float S = 0.f, Q = 0.f;
#pragma unroll
    for (int k = 0; k < 8; ++k) { S += red[row * 8 + k]; Q += red[256 + row * 8 + k]; }
    float mean = S * 0.015625f;
    float var  = Q * 0.015625f - mean * mean;
    stats[row] = mean;
    stats[32 + row] = rsqrtf(var + 1e-5f);
  }
  __syncthreads();
  float mean = stats[row], inv = stats[32 + row];
#pragma unroll
  for (int k = 0; k < 8; ++k) {
    int c0 = s * 8 + k;
    float x = hf[row * 64 + c0];
    dst[row * 64 + c0] = bf16bits((x - mean) * inv * gamma[c0]);
  }
}

// ---------------------------------------------------------------------------
// Fused kernel: 1 block = 32 edges, 256 threads (8 waves).
// Dynamic LDS layout (bytes), r-buffer aliases dead early-phase buffers:
//   rbuf   @ 0       : 32*768 f32 = 98304   (GEMM3 output)
//     hf   @ 0       : 32*64  f32 = 8192    (alias; SiLU outputs, dead pre-GEMM3)
//     featb@ 8192    : 32*32  bf16= 2048    (alias; dead after GEMM1)
//     h1b  @ 10240   : 32*64  bf16= 4096    (alias; dead after GEMM2)
//   h2b    @ 98304   : 32*64  bf16= 4096    (live during GEMM3 -> NOT aliased)
//   basis_s@ 102400  : 32*27  f32 = 3456
//   red    @ 105856  : 512    f32 = 2048
//   stats  @ 107904  : 64     f32 = 256
// total 108160 B
// ---------------------------------------------------------------------------
#define SMEM_BYTES 108160

__global__ __launch_bounds__(256)
void pairwise_tp_fused(const float* __restrict__ feat,
                       const float* __restrict__ basis,
                       const float* __restrict__ b1, const float* __restrict__ g1,
                       const float* __restrict__ b2, const float* __restrict__ g2,
                       const float* __restrict__ b3,
                       const unsigned short* __restrict__ wt,
                       float* __restrict__ out) {
  extern __shared__ char smem[];
  float*          rbuf    = (float*)smem;
  float*          hf      = (float*)smem;
  unsigned short* featb   = (unsigned short*)(smem + 8192);
  unsigned short* h1b     = (unsigned short*)(smem + 10240);
  unsigned short* h2b     = (unsigned short*)(smem + 98304);
  float*          basis_s = (float*)(smem + 102400);
  float*          red     = (float*)(smem + 105856);
  float*          stats   = (float*)(smem + 107904);

  const unsigned short* w1t = wt;           // [64][32]
  const unsigned short* w2t = wt + 2048;    // [64][64]
  const unsigned short* w3t = wt + 6144;    // [768][64]

  const int t    = threadIdx.x;
  const int wv   = t >> 5;
  const int lane = t & 31;
  const int hh   = lane >> 4;   // lane half
  const int lm   = lane & 15;
  const int edge0 = blockIdx.x * 32;

  // ---- stage feat (bf16, K padded 17->32) and basis ----
  for (int i = t; i < 32 * 32; i += 256) {
    int r = i >> 5, c = i & 31;
    float v = (c < 17) ? feat[(size_t)(edge0 + r) * 17 + c] : 0.0f;
    featb[i] = bf16bits(v);
  }
  for (int i = t; i < 32 * 27; i += 256)
    basis_s[i] = basis[(size_t)edge0 * 27 + i];
  __syncthreads();

  // ---- GEMM1: (32x32) @ (32x64) + b1, SiLU -> hf ----
  {
    int mt = wv >> 2, nt = wv & 3;
    Frag a = load_frag(featb, mt * 16 + lm, 32, 0, hh);
    Frag b = load_frag(w1t,   nt * 16 + lm, 32, 0, hh);
    v8f c = {};
    c = __builtin_amdgcn_wmma_f32_16x16x32_bf16(false, a.v, false, b.v,
                                                (short)0, c, false, false);
    int colg = nt * 16 + lm;
    float bias = b1[colg];
#pragma unroll
    for (int vi = 0; vi < 8; ++vi) {
      float x = c[vi] + bias;
      hf[(mt * 16 + vi + 8 * hh) * 64 + colg] = silu(x);
    }
  }
  __syncthreads();
  ln_block(hf, h1b, g1, red, stats, t);
  __syncthreads();

  // ---- GEMM2: (32x64) @ (64x64) + b2, SiLU -> hf ----
  {
    int mt = wv >> 2, nt = wv & 3;
    v8f c = {};
#pragma unroll
    for (int ks = 0; ks < 2; ++ks) {
      Frag a = load_frag(h1b, mt * 16 + lm, 64, ks * 32, hh);
      Frag b = load_frag(w2t, nt * 16 + lm, 64, ks * 32, hh);
      c = __builtin_amdgcn_wmma_f32_16x16x32_bf16(false, a.v, false, b.v,
                                                  (short)0, c, false, false);
    }
    int colg = nt * 16 + lm;
    float bias = b2[colg];
#pragma unroll
    for (int vi = 0; vi < 8; ++vi) {
      float x = c[vi] + bias;
      hf[(mt * 16 + vi + 8 * hh) * 64 + colg] = silu(x);
    }
  }
  __syncthreads();
  ln_block(hf, h2b, g2, red, stats, t);
  __syncthreads();

  // ---- GEMM3: (32x64) @ (64x768) + b3 -> rbuf (fp32, LDS) ----
  for (int tile = wv; tile < 96; tile += 8) {
    int mt = tile & 1, nt = tile >> 1;
    v8f c = {};
#pragma unroll
    for (int ks = 0; ks < 2; ++ks) {
      Frag a = load_frag(h2b, mt * 16 + lm, 64, ks * 32, hh);
      Frag b = load_frag(w3t, nt * 16 + lm, 64, ks * 32, hh);
      c = __builtin_amdgcn_wmma_f32_16x16x32_bf16(false, a.v, false, b.v,
                                                  (short)0, c, false, false);
    }
    int colg = nt * 16 + lm;
    float bias = b3[colg];
#pragma unroll
    for (int vi = 0; vi < 8; ++vi)
      rbuf[(mt * 16 + vi + 8 * hh) * 768 + colg] = c[vi] + bias;
  }
  __syncthreads();

  // ---- Tensor product: out[e][o*3+d][i*3+m] = sum_f R[o,i,f]*Bs[d,m,f] ----
  // 32 edges * 48 out-rows = 1536 rows; 6 rows/thread; ds_load_b128 reads,
  // b128 global stores overlapping the HBM store stream (the bottleneck).
#pragma unroll
  for (int rr = 0; rr < 6; ++rr) {
    int rowid = rr * 256 + t;
    int e  = rowid / 48;
    int ro = rowid - e * 48;
    int o  = ro / 3;
    int d  = ro - o * 3;
    const float* bs = basis_s + e * 27 + d * 9;
    float b00 = bs[0], b01 = bs[1], b02 = bs[2];
    float b10 = bs[3], b11 = bs[4], b12 = bs[5];
    float b20 = bs[6], b21 = bs[7], b22 = bs[8];

    // R row segment is 16B-aligned (o*192 bytes): read as 12 x ds_load_b128.
    const float* rp = rbuf + e * 768 + o * 48;
    float rv[48];
#pragma unroll
    for (int j = 0; j < 12; ++j) {
      f32x4 q = *(const f32x4*)(rp + 4 * j);
      rv[4 * j + 0] = q[0];
      rv[4 * j + 1] = q[1];
      rv[4 * j + 2] = q[2];
      rv[4 * j + 3] = q[3];
    }

    float acc[48];
#pragma unroll
    for (int i = 0; i < 16; ++i) {
      float r0 = rv[i * 3 + 0], r1 = rv[i * 3 + 1], r2 = rv[i * 3 + 2];
      acc[i * 3 + 0] = r0 * b00 + r1 * b01 + r2 * b02;
      acc[i * 3 + 1] = r0 * b10 + r1 * b11 + r2 * b12;
      acc[i * 3 + 2] = r0 * b20 + r1 * b21 + r2 * b22;
    }
    float* gp = out + (size_t)(edge0 + e) * 2304 + (size_t)ro * 48;
#pragma unroll
    for (int j = 0; j < 12; ++j) {
      f32x4 v4 = { acc[4 * j], acc[4 * j + 1], acc[4 * j + 2], acc[4 * j + 3] };
      *(f32x4*)(gp + 4 * j) = v4;
    }
  }
}

extern "C" void kernel_launch(void* const* d_in, const int* in_sizes, int n_in,
                              void* d_out, int out_size, void* d_ws, size_t ws_size,
                              hipStream_t stream) {
  const float* feat  = (const float*)d_in[0];
  const float* basis = (const float*)d_in[1];
  const float* W1    = (const float*)d_in[2];
  const float* b1    = (const float*)d_in[3];
  const float* g1    = (const float*)d_in[4];
  const float* W2    = (const float*)d_in[5];
  const float* b2    = (const float*)d_in[6];
  const float* g2    = (const float*)d_in[7];
  const float* W3    = (const float*)d_in[8];
  const float* b3    = (const float*)d_in[9];
  unsigned short* wt = (unsigned short*)d_ws;   // needs 110592 B

  (void)hipFuncSetAttribute((const void*)pairwise_tp_fused,
                            hipFuncAttributeMaxDynamicSharedMemorySize,
                            SMEM_BYTES);

  // 55296 weight elements total -> 216 blocks of 256
  prep_weights<<<216, 256, 0, stream>>>(W1, W2, W3, wt);

  // 32768 edges / 32 per block = 1024 blocks
  pairwise_tp_fused<<<1024, 256, SMEM_BYTES, stream>>>(
      feat, basis, b1, g1, b2, g2, b3, wt, (float*)d_out);
}
